// _GatedMessagePassingLayer_39685497815178
// MI455X (gfx1250) — matrix operand order repaired and checked by reading, not compiled, not run on hardware
//
#include <hip/hip_runtime.h>
#include <hip/hip_bf16.h>
#include <math.h>

// ---------------------------------------------------------------------------
// Gated message-passing layer, fused, for MI455X (gfx1250, wave32, WMMA).
// B=32, N=128, STATE=128, EDGE=32, INNER=256, pair-dim = 288.
// Compute-bound (~1.9e11 FLOPs vs ~70MB traffic) -> bf16 WMMA w/ f32 accum.
// ---------------------------------------------------------------------------

typedef __attribute__((ext_vector_type(16))) __bf16 v16bf;
typedef __attribute__((ext_vector_type(8)))  __bf16 v8bf;
typedef __attribute__((ext_vector_type(8)))  float  v8f;

#define BDIM 32
#define NDIM 128
#define SDIM 128
#define EDIM 32
#define PDIM 288     // 2*SDIM + EDIM
#define HDIM 256     // INNER

// LDS layout (bytes)
#define PAIR_STRIDE 296                         // elems, padded
#define HM_STRIDE   264                         // elems, padded
#define PAIR_BYTES  (128 * PAIR_STRIDE * 2)     // 75776
#define HM_OFF      PAIR_BYTES
#define HM_BYTES    (128 * HM_STRIDE * 2)       // 67584
#define GLIN_OFF    (HM_OFF + HM_BYTES)         // 143360
#define GATE_OFF    (GLIN_OFF + 512)
#define AGG_OFF     (GATE_OFF + 512)
#define HROW_OFF    (AGG_OFF + 512)
#define MASK_OFF    (HROW_OFF + 512)
#define HID_OFF     (MASK_OFF + 512)
#define XROW_OFF    (HID_OFF + 1024)
#define RED_OFF     (XROW_OFF + 512)
#define SMEM_BYTES  (RED_OFF + 32)              // 147488

// packed-weight sizes (elements) in d_ws
#define WM1P_ELEMS (PDIM * HDIM)   // 73728  (9 K-tiles x 16 N-tiles)
#define WG1P_ELEMS (PDIM * HDIM)   // 73728
#define WM2P_ELEMS (HDIM * SDIM)   // 32768  (8 K-tiles x 8 N-tiles)
#define PACK_TOTAL (WM1P_ELEMS + WG1P_ELEMS + WM2P_ELEMS)  // 180224

// ---------------------------------------------------------------------------
// Pre-kernel: convert f32 weights to bf16 in WMMA B-fragment order.
// Block of K=32 x N=16: elem (lane, t) = W[kt*32 + (lane>>4)*16 + t][nt*16 + (lane&15)]
// Blocks ordered nt-major so the kt inner loop streams contiguously.
// ---------------------------------------------------------------------------
__global__ void pack_weights_kernel(const float* __restrict__ Wm1,
                                    const float* __restrict__ Wg1,
                                    const float* __restrict__ Wm2,
                                    __bf16* __restrict__ ws) {
  int idx = blockIdx.x * 256 + threadIdx.x;
  if (idx >= PACK_TOTAL) return;
  const float* W;
  __bf16* dst;
  int e, nKt, Ncols;
  if (idx < WM1P_ELEMS)                 { W = Wm1; dst = ws;                            e = idx;                           nKt = 9; Ncols = 256; }
  else if (idx < WM1P_ELEMS+WG1P_ELEMS) { W = Wg1; dst = ws + WM1P_ELEMS;               e = idx - WM1P_ELEMS;              nKt = 9; Ncols = 256; }
  else                                  { W = Wm2; dst = ws + WM1P_ELEMS + WG1P_ELEMS;  e = idx - WM1P_ELEMS - WG1P_ELEMS; nKt = 8; Ncols = 128; }
  int bidx = e >> 9, r = e & 511, lane = r >> 4, t = r & 15;
  int nt = bidx / nKt, kt = bidx % nKt;
  int k = kt * 32 + (lane >> 4) * 16 + t;
  int n = nt * 16 + (lane & 15);
  dst[e] = (__bf16)W[k * Ncols + n];
}

// A fragment (16x32 bf16) from an LDS tile, in this kernel's contiguous k order (same as packed B):
// lane l: M = l&15 ; elems 0..15 -> K = kbase + 16*(l>>4) + t.
__device__ inline v16bf load_a_frag(const __bf16* tile, int stride, int row,
                                    int kbase, int lh) {
  union { v16bf v; v8bf h[2]; } u;
  const __bf16* p = tile + row * stride + kbase + 16 * lh;
  u.h[0] = *(const v8bf*)(p);
  u.h[1] = *(const v8bf*)(p + 8);
  return u.v;
}

// ---------------------------------------------------------------------------
// Main kernel: one block per (b, i). 256 threads = 8 waves; wave w owns rows
// j in [16w, 16w+16).
// ---------------------------------------------------------------------------
__global__ __launch_bounds__(256)
void gmp_main_kernel(const float* __restrict__ h,
                     const float* __restrict__ edge_feat,
                     const float* __restrict__ edge_mask,
                     const float* __restrict__ bm1,
                     const float* __restrict__ bm2,
                     const float* __restrict__ bg1,
                     const float* __restrict__ Wg2,
                     const float* __restrict__ bg2,
                     const float* __restrict__ Wu1,
                     const float* __restrict__ bu1,
                     const float* __restrict__ Wu2,
                     const float* __restrict__ bu2,
                     const float* __restrict__ ln_w,
                     const float* __restrict__ ln_b,
                     const __bf16* __restrict__ Wm1p,
                     const __bf16* __restrict__ Wg1p,
                     const __bf16* __restrict__ Wm2p,
                     float* __restrict__ out) {
  extern __shared__ char smem[];
  __bf16* pairA   = (__bf16*)(smem);
  __bf16* Hm      = (__bf16*)(smem + HM_OFF);
  float*  gateLin = (float*)(smem + GLIN_OFF);
  float*  gateV   = (float*)(smem + GATE_OFF);
  float*  aggL    = (float*)(smem + AGG_OFF);
  float*  hrow    = (float*)(smem + HROW_OFF);
  float*  maskRow = (float*)(smem + MASK_OFF);
  float*  hid     = (float*)(smem + HID_OFF);
  float*  xrow    = (float*)(smem + XROW_OFF);
  float*  red     = (float*)(smem + RED_OFF);

  const int tid  = threadIdx.x;
  const int wave = tid >> 5;
  const int lane = tid & 31;
  const int lh   = lane >> 4;   // lane half
  const int nc   = lane & 15;   // column-in-tile / row-in-A-tile
  const int bb   = blockIdx.x / NDIM;
  const int ii   = blockIdx.x % NDIM;
  const int mbase = wave * 16;

  // ---- phase 0: h_i row and mask row ----
  if (tid < SDIM) {
    hrow[tid]    = h[(size_t)(bb * NDIM + ii) * SDIM + tid];
    maskRow[tid] = edge_mask[(size_t)(bb * NDIM + ii) * NDIM + tid];
  }
  __syncthreads();

  // ---- phase 1: build pair tile [128 x 288] bf16 in LDS ----
  for (int e = tid; e < NDIM * PDIM; e += 256) {
    int j = e / PDIM, k = e - j * PDIM;
    float v;
    if (k < SDIM)           v = hrow[k];
    else if (k < 2 * SDIM)  v = h[(size_t)(bb * NDIM + j) * SDIM + (k - SDIM)];
    else                    v = edge_feat[((size_t)(bb * NDIM + ii) * NDIM + j) * EDIM + (k - 2 * SDIM)];
    pairA[j * PAIR_STRIDE + k] = (__bf16)v;
  }
  __syncthreads();

  // ---- phase 2: GEMM1 (K=288): msg hidden -> Hm (bf16 LDS), gate hidden
  //      folded into per-row dot with Wg2. Shared A fragments. ----
  float gpart[8];
#pragma unroll
  for (int v = 0; v < 8; ++v) gpart[v] = 0.f;

  for (int nt = 0; nt < 16; ++nt) {
    v8f accM = {};
    v8f accG = {};
    const __bf16* bpm = Wm1p + (nt * 9) * 512 + lane * 16;
    const __bf16* bpg = Wg1p + (nt * 9) * 512 + lane * 16;
#pragma unroll
    for (int kt = 0; kt < 9; ++kt) {
      __builtin_prefetch(bpm + (kt + 2) * 512, 0, 0);
      v16bf a  = load_a_frag(pairA, PAIR_STRIDE, mbase + nc, kt * 32, lh);
      v16bf bM = *(const v16bf*)(bpm + kt * 512);
      v16bf bG = *(const v16bf*)(bpg + kt * 512);
      accM = __builtin_amdgcn_wmma_f32_16x16x32_bf16(false, a, false, bM, (short)0, accM, false, false);
      accG = __builtin_amdgcn_wmma_f32_16x16x32_bf16(false, a, false, bG, (short)0, accG, false, false);
    }
    const int col = nt * 16 + nc;
    const float b1m = bm1[col], b1g = bg1[col], w2 = Wg2[col];
#pragma unroll
    for (int v = 0; v < 8; ++v) {
      const int row = mbase + v + 8 * lh;
      Hm[row * HM_STRIDE + col] = (__bf16)tanhf(accM[v] + b1m);
      gpart[v] += tanhf(accG[v] + b1g) * w2;
    }
  }
  // reduce gate dot across the 16 lanes that share a row
#pragma unroll
  for (int m = 1; m < 16; m <<= 1) {
#pragma unroll
    for (int v = 0; v < 8; ++v) gpart[v] += __shfl_xor(gpart[v], m, 32);
  }
  if (nc == 0) {
#pragma unroll
    for (int v = 0; v < 8; ++v) gateLin[mbase + v + 8 * lh] = gpart[v];
  }
  if (tid < SDIM) aggL[tid] = 0.f;
  __syncthreads();

  // ---- phase 3: gate = sigmoid(lin + bg2) * mask; denom & degree ----
  if (tid < NDIM) {
    float g = 1.f / (1.f + __expf(-(gateLin[tid] + bg2[0])));
    gateV[tid] = g * maskRow[tid];
  }
  __syncthreads();
  if (wave == 0) {
    float s = 0.f, d = 0.f;
#pragma unroll
    for (int t = 0; t < 4; ++t) {
      s += gateV[lane * 4 + t];
      d += maskRow[lane * 4 + t];
    }
#pragma unroll
    for (int m = 1; m < 32; m <<= 1) {
      s += __shfl_xor(s, m, 32);
      d += __shfl_xor(d, m, 32);
    }
    if (lane == 0) {
      red[0] = 1.f / fmaxf(s, 1.f);        // 1/denom
      red[1] = d * (1.f / (float)(NDIM - 1));  // degree_norm
    }
  }

  // ---- phase 4: GEMM2 (K=256, N=128): msg = Hm @ Wm2 + bm2, gated agg ----
  for (int nt = 0; nt < 8; ++nt) {
    v8f acc = {};
    const __bf16* bp = Wm2p + (nt * 8) * 512 + lane * 16;
#pragma unroll
    for (int kt = 0; kt < 8; ++kt) {
      v16bf a = load_a_frag(Hm, HM_STRIDE, mbase + nc, kt * 32, lh);
      v16bf b = *(const v16bf*)(bp + kt * 512);
      acc = __builtin_amdgcn_wmma_f32_16x16x32_bf16(false, a, false, b, (short)0, acc, false, false);
    }
    const int col = nt * 16 + nc;
    const float b2 = bm2[col];
    float p = 0.f;
#pragma unroll
    for (int v = 0; v < 8; ++v) {
      const int row = mbase + v + 8 * lh;
      p += gateV[row] * (acc[v] + b2);
    }
    atomicAdd(&aggL[col], p);
  }
  __syncthreads();

  // ---- phase 5: agg normalize ----
  const float invDenom = red[0];
  const float degNorm  = red[1];
  if (tid < SDIM) aggL[tid] *= invDenom;
  __syncthreads();

  // ---- phase 6: update MLP hidden (257 -> 256), one hidden unit / thread ----
  {
    float s = bu1[tid];
#pragma unroll 4
    for (int k = 0; k < SDIM; ++k) s += hrow[k] * Wu1[k * HDIM + tid];
#pragma unroll 4
    for (int k = 0; k < SDIM; ++k) s += aggL[k] * Wu1[(SDIM + k) * HDIM + tid];
    s += degNorm * Wu1[2 * SDIM * HDIM + tid];
    hid[tid] = tanhf(s);
  }
  __syncthreads();

  // ---- phase 7: delta (256 -> 128), x = h + tanh(delta) ----
  if (tid < SDIM) {
    float s = bu2[tid];
#pragma unroll 4
    for (int n = 0; n < HDIM; ++n) s += hid[n] * Wu2[n * SDIM + tid];
    xrow[tid] = hrow[tid] + tanhf(s);
  }
  __syncthreads();

  // ---- phase 8: layernorm ----
  if (wave == 0) {
    float s = 0.f, q = 0.f;
#pragma unroll
    for (int t = 0; t < 4; ++t) {
      float x = xrow[lane * 4 + t];
      s += x; q += x * x;
    }
#pragma unroll
    for (int m = 1; m < 32; m <<= 1) {
      s += __shfl_xor(s, m, 32);
      q += __shfl_xor(q, m, 32);
    }
    if (lane == 0) {
      float mu  = s * (1.f / (float)SDIM);
      float var = q * (1.f / (float)SDIM) - mu * mu;
      red[2] = mu;
      red[3] = rsqrtf(var + 1e-5f);
    }
  }
  __syncthreads();
  if (tid < SDIM) {
    float x = (xrow[tid] - red[2]) * red[3];
    out[(size_t)(bb * NDIM + ii) * SDIM + tid] = x * ln_w[tid] + ln_b[tid];
  }
}

// ---------------------------------------------------------------------------
extern "C" void kernel_launch(void* const* d_in, const int* in_sizes, int n_in,
                              void* d_out, int out_size, void* d_ws, size_t ws_size,
                              hipStream_t stream) {
  const float* h         = (const float*)d_in[0];
  const float* edge_feat = (const float*)d_in[1];
  const float* edge_mask = (const float*)d_in[2];
  const float* Wm1 = (const float*)d_in[3];
  const float* bm1 = (const float*)d_in[4];
  const float* Wm2 = (const float*)d_in[5];
  const float* bm2 = (const float*)d_in[6];
  const float* Wg1 = (const float*)d_in[7];
  const float* bg1 = (const float*)d_in[8];
  const float* Wg2 = (const float*)d_in[9];
  const float* bg2 = (const float*)d_in[10];
  const float* Wu1 = (const float*)d_in[11];
  const float* bu1 = (const float*)d_in[12];
  const float* Wu2 = (const float*)d_in[13];
  const float* bu2 = (const float*)d_in[14];
  const float* lnw = (const float*)d_in[15];
  const float* lnb = (const float*)d_in[16];
  float* out = (float*)d_out;

  __bf16* wsp = (__bf16*)d_ws;   // needs PACK_TOTAL*2 = 360,448 bytes

  (void)in_sizes; (void)n_in; (void)out_size; (void)ws_size;

  hipFuncSetAttribute((const void*)gmp_main_kernel,
                      hipFuncAttributeMaxDynamicSharedMemorySize,
                      (int)SMEM_BYTES);

  pack_weights_kernel<<<(PACK_TOTAL + 255) / 256, 256, 0, stream>>>(Wm1, Wg1, Wm2, wsp);

  gmp_main_kernel<<<BDIM * NDIM, 256, SMEM_BYTES, stream>>>(
      h, edge_feat, edge_mask, bm1, bm2, bg1, Wg2, bg2,
      Wu1, bu1, Wu2, bu2, lnw, lnb,
      wsp, wsp + WM1P_ELEMS, wsp + WM1P_ELEMS + WG1P_ELEMS, out);
}
